// SparseGate_2302102471007
// MI455X (gfx1250) — matrix operand order repaired and checked by reading, not compiled, not run on hardware
//
#include <hip/hip_runtime.h>
#include <hip/hip_bf16.h>

typedef __attribute__((ext_vector_type(16))) __bf16 v16bf;
typedef __attribute__((ext_vector_type(4)))  __bf16 v4bf;
typedef __attribute__((ext_vector_type(8)))  float  v8f;
typedef __attribute__((ext_vector_type(4)))  int    v4i;

#define TOKENS 16384
#define DDIM   4096
#define NEXP   256
#define TOPK   8
#define TOK_PER_WG 32
#define NWG    (TOKENS / TOK_PER_WG)   // 512 workgroups
#define KC     256                     // K-chunk staged in LDS
#define NOISE_STD 0.1f
#define EPS 1e-6f

union Frag {
    v16bf bf;
    v4i   q[2];
};

__device__ __forceinline__ void pack4(v16bf& v, int o, const float4& f) {
    v[o + 0] = (__bf16)f.x;
    v[o + 1] = (__bf16)f.y;
    v[o + 2] = (__bf16)f.z;
    v[o + 3] = (__bf16)f.w;
}

// ---------------------------------------------------------------------------
// Kernel 0: one-time W f32 -> bf16 conversion into workspace (4MB -> 2MB).
// ---------------------------------------------------------------------------
__global__ __launch_bounds__(512)
void cvt_w(const float* __restrict__ W, __bf16* __restrict__ Wbf) {
    const size_t i = ((size_t)blockIdx.x * 512 + threadIdx.x) * 4;
    float4 f = *(const float4*)(W + i);
    v4bf o;
    o[0] = (__bf16)f.x; o[1] = (__bf16)f.y; o[2] = (__bf16)f.z; o[3] = (__bf16)f.w;
    *(v4bf*)(Wbf + i) = o;
}

// ---------------------------------------------------------------------------
// Main kernel: one workgroup = 32 tokens x 256 experts. 512 threads = 16 waves.
// Wave w owns experts [16w,16w+16) and two M-tiles (tokens 0-15, 16-31).
// x tile staged in LDS as bf16 (converted once); W pre-converted to bf16.
// ---------------------------------------------------------------------------
__global__ __launch_bounds__(512)
void moe_gate_main(const float*  __restrict__ x,      // [TOKENS][DDIM]
                   const float*  __restrict__ noise,  // [TOKENS][NEXP]
                   const __bf16* __restrict__ Wbf,    // [NEXP][DDIM] bf16
                   const float*  __restrict__ bias,   // [NEXP]
                   float* __restrict__ gates,         // [TOKENS][NEXP]
                   int*   __restrict__ topIdx,        // [TOKENS][TOPK]
                   float* __restrict__ wsImp,         // [NWG][NEXP]
                   float* __restrict__ wsLoad)        // [NWG][NEXP]
{
    // padded row stride: 264 bf16 = 132 dwords == 4 mod 64 banks -> conflict-free
    __shared__ __align__(16) __bf16 xs[TOK_PER_WG][KC + 8];
    __shared__ float lgt[TOK_PER_WG][NEXP];     // logits, later overwritten by probs
    __shared__ float topgL[TOK_PER_WG][TOPK];
    __shared__ int   topiL[TOK_PER_WG][TOPK];

    const int tid   = threadIdx.x;
    const int lane  = tid & 31;
    const int wave  = tid >> 5;
    const int wg    = blockIdx.x;
    const int tBase = wg * TOK_PER_WG;

    const int eBase = wave * 16;
    const int half  = lane >> 4;      // 0 | 1
    const int l15   = lane & 15;
    const int aKoff = half * 8;       // A frag: lanes0-15 K{0..7,16..23}; 16-31 K{8..15,24..31}
    const int bKoff = half * 8;       // B frag: lane=column, K{0..7,16..23} / K{8..15,24..31}

    // staging assignment: each thread converts 16 floats of one row
    const int srow = tid >> 4;
    const int scol = (tid & 15) << 4;
    const float* srcBase = x + (size_t)(tBase + srow) * DDIM + scol;

    const __bf16* bColBase = Wbf + (size_t)(eBase + l15) * DDIM + bKoff;

    v8f acc0 = {};   // tokens 0-15
    v8f acc1 = {};   // tokens 16-31

    for (int chunk = 0; chunk < DDIM / KC; ++chunk) {
        // ---- cooperative stage: 32x256 f32 -> bf16 into LDS (each elem once)
        {
            const float4* s = (const float4*)(srcBase + (size_t)chunk * KC);
            float4 f0 = s[0], f1 = s[1], f2 = s[2], f3 = s[3];
            Frag u;
            pack4(u.bf, 0, f0); pack4(u.bf, 4, f1);
            pack4(u.bf, 8, f2); pack4(u.bf, 12, f3);
            *(v4i*)&xs[srow][scol]     = u.q[0];
            *(v4i*)&xs[srow][scol + 8] = u.q[1];
        }
        __syncthreads();

        // ---- 8 k-steps of WMMA per chunk; B frag shared by both M-tiles
#pragma unroll 2
        for (int ks = 0; ks < KC; ks += 32) {
            Frag bfr, af0, af1;
            const v4i* bp = (const v4i*)(bColBase + (size_t)chunk * KC + ks);
            bfr.q[0] = bp[0];
            bfr.q[1] = bp[2];
            af0.q[0] = *(const v4i*)&xs[l15][ks + aKoff];
            af0.q[1] = *(const v4i*)&xs[l15][ks + aKoff + 16];
            af1.q[0] = *(const v4i*)&xs[16 + l15][ks + aKoff];
            af1.q[1] = *(const v4i*)&xs[16 + l15][ks + aKoff + 16];

            acc0 = __builtin_amdgcn_wmma_f32_16x16x32_bf16(
                false, af0.bf, false, bfr.bf, (short)0, acc0, false, false);
            acc1 = __builtin_amdgcn_wmma_f32_16x16x32_bf16(
                false, af1.bf, false, bfr.bf, (short)0, acc1, false, false);
        }
        __syncthreads();
    }

    // ---- epilogue stage 1: logits (+bias +0.1*noise) into LDS
    {
        const float bv = bias[eBase + l15];
#pragma unroll
        for (int r = 0; r < 8; ++r) {
            const int m0 = (half << 3) + r;          // tokens 0-15 tile
            const int m1 = 16 + m0;                  // tokens 16-31 tile
            lgt[m0][eBase + l15] = acc0[r] + bv
                + NOISE_STD * noise[(size_t)(tBase + m0) * NEXP + eBase + l15];
            lgt[m1][eBase + l15] = acc1[r] + bv
                + NOISE_STD * noise[(size_t)(tBase + m1) * NEXP + eBase + l15];
        }
    }
    __syncthreads();

    // ---- epilogue stage 2: per-token top-8 / softmax; wave handles 2 tokens
    for (int ti = 0; ti < 2; ++ti) {
        const int t  = 2 * wave + ti;
        const int eb = lane * 8;
        float v[8], lv[8];
#pragma unroll
        for (int j = 0; j < 8; ++j) { v[j] = lgt[t][eb + j]; lv[j] = v[j]; }

        float topv[TOPK];
        int   topi8[TOPK];
#pragma unroll
        for (int it = 0; it < TOPK; ++it) {
            float bvv = lv[0]; int bi = eb;
#pragma unroll
            for (int j = 1; j < 8; ++j)
                if (lv[j] > bvv) { bvv = lv[j]; bi = eb + j; }
            for (int off = 16; off > 0; off >>= 1) {      // ties -> lowest index
                float ov = __shfl_xor(bvv, off, 32);
                int   oi = __shfl_xor(bi,  off, 32);
                if (ov > bvv || (ov == bvv && oi < bi)) { bvv = ov; bi = oi; }
            }
            topv[it] = bvv; topi8[it] = bi;
            if ((bi >> 3) == lane) lv[bi & 7] = -__builtin_inff();
        }

        const float tmax = topv[0];                       // row max
        float g[TOPK], tsum = 0.f;
#pragma unroll
        for (int k = 0; k < TOPK; ++k) { g[k] = __expf(topv[k] - tmax); tsum += g[k]; }
        const float ginv = 1.f / tsum;

        // fully-written gates row (zeros elsewhere)
        float gv[8];
#pragma unroll
        for (int j = 0; j < 8; ++j) {
            float out = 0.f;
            const int e = eb + j;
#pragma unroll
            for (int k = 0; k < TOPK; ++k)
                if (topi8[k] == e) out = g[k] * ginv;
            gv[j] = out;
        }
        const size_t gBase = (size_t)(tBase + t) * NEXP + eb;
        *(float4*)(gates + gBase)     = make_float4(gv[0], gv[1], gv[2], gv[3]);
        *(float4*)(gates + gBase + 4) = make_float4(gv[4], gv[5], gv[6], gv[7]);

        if (lane == 0) {
#pragma unroll
            for (int k = 0; k < TOPK; ++k) {
                topIdx[(size_t)(tBase + t) * TOPK + k] = topi8[k];
                topgL[t][k] = g[k] * ginv;
                topiL[t][k] = topi8[k];
            }
        }

        // full-row softmax probs -> overwrite lgt row
        float s = 0.f;
#pragma unroll
        for (int j = 0; j < 8; ++j) s += __expf(v[j] - tmax);
        for (int off = 16; off > 0; off >>= 1) s += __shfl_xor(s, off, 32);
        const float sinv = 1.f / s;
#pragma unroll
        for (int j = 0; j < 8; ++j) lgt[t][eb + j] = __expf(v[j] - tmax) * sinv;
    }
    __syncthreads();

    // ---- deterministic per-WG partials
    if (tid < NEXP) {
        float sl = 0.f;
#pragma unroll 4
        for (int r = 0; r < TOK_PER_WG; ++r) sl += lgt[r][tid];
        float si = 0.f;
        for (int r = 0; r < TOK_PER_WG; ++r)
#pragma unroll
            for (int k = 0; k < TOPK; ++k)
                if (topiL[r][k] == tid) si += topgL[r][k];
        wsLoad[(size_t)wg * NEXP + tid] = sl;
        wsImp [(size_t)wg * NEXP + tid] = si;
    }
}

// ---------------------------------------------------------------------------
// Final scalar: CV(importance) + CV(load), std with ddof=1.
// ---------------------------------------------------------------------------
__global__ __launch_bounds__(256)
void moe_gate_loss(const float* __restrict__ wsImp,
                   const float* __restrict__ wsLoad,
                   float* __restrict__ lossOut)
{
    __shared__ float imp[NEXP], ld[NEXP];
    const int e = threadIdx.x;
    float si = 0.f, sl = 0.f;
    for (int w = 0; w < NWG; ++w) {
        si += wsImp [(size_t)w * NEXP + e];
        sl += wsLoad[(size_t)w * NEXP + e];
    }
    imp[e] = si; ld[e] = sl;
    __syncthreads();

    if (e == 0) {
        double mi = 0.0, ml = 0.0;
        for (int i = 0; i < NEXP; ++i) { mi += imp[i]; ml += ld[i]; }
        mi /= NEXP; ml /= NEXP;
        double vi = 0.0, vl = 0.0;
        for (int i = 0; i < NEXP; ++i) {
            double di = imp[i] - mi; vi += di * di;
            double dl = ld[i]  - ml; vl += dl * dl;
        }
        vi /= (NEXP - 1); vl /= (NEXP - 1);
        double loss = sqrt(vi) / (mi + (double)EPS) + sqrt(vl) / (ml + (double)EPS);
        *lossOut = (float)loss;
    }
}

extern "C" void kernel_launch(void* const* d_in, const int* in_sizes, int n_in,
                              void* d_out, int out_size, void* d_ws, size_t ws_size,
                              hipStream_t stream) {
    const float* x     = (const float*)d_in[0];   // [16384,4096]
    const float* noise = (const float*)d_in[1];   // [16384,256]
    const float* Wm    = (const float*)d_in[2];   // [256,4096]
    const float* bias  = (const float*)d_in[3];   // [256]

    float* out    = (float*)d_out;
    float* gates  = out;                                            // 16384*256 f32
    int*   topIdx = (int*)(out + (size_t)TOKENS * NEXP);            // 16384*8 i32
    float* loss   = out + (size_t)TOKENS * NEXP + (size_t)TOKENS * TOPK;

    __bf16* Wbf   = (__bf16*)d_ws;                                  // 256*4096 bf16 (2MB)
    float* wsImp  = (float*)((char*)d_ws + (size_t)NEXP * DDIM * sizeof(__bf16));
    float* wsLoad = wsImp + (size_t)NWG * NEXP;

    cvt_w<<<(NEXP * DDIM) / (512 * 4), 512, 0, stream>>>(Wm, Wbf);
    moe_gate_main<<<NWG, 512, 0, stream>>>(x, noise, Wbf, bias,
                                           gates, topIdx, wsImp, wsLoad);
    moe_gate_loss<<<1, 256, 0, stream>>>(wsImp, wsLoad, loss);
}
